// ExampleModel_59055800320697
// MI455X (gfx1250) — matrix operand order repaired and checked by reading, not compiled, not run on hardware
//
#include <hip/hip_runtime.h>

#define T_TOK 8192
#define DDIM  512
#define HDIM  2048
#define NEXP  8

typedef __attribute__((ext_vector_type(16))) __bf16 v16bf;
typedef __attribute__((ext_vector_type(8)))  float  v8f;

static __device__ __forceinline__ unsigned short f2bf(float f) {
  unsigned u = __float_as_uint(f);
  u += 0x7fffu + ((u >> 16) & 1u);            // round-to-nearest-even
  return (unsigned short)(u >> 16);
}
static __device__ __forceinline__ unsigned pk2(float a, float b) {
  return (unsigned)f2bf(a) | ((unsigned)f2bf(b) << 16);
}
struct B32x8 { uint4 lo, hi; };
static __device__ __forceinline__ v16bf frag_from(const uint4* p, int i0, int i1) {
  B32x8 t; t.lo = p[i0]; t.hi = p[i1];
  return __builtin_bit_cast(v16bf, t);
}
static __device__ __forceinline__ float gelu_exact(float x) {
  return 0.5f * x * (1.0f + erff(x * 0.70710678118654752f));
}

// CDNA5 async global->LDS copy (ASYNCcnt-tracked, no VGPR data round-trip).
static __device__ __forceinline__ void async_ld_b128(unsigned lds_off, const void* gptr) {
  asm volatile("global_load_async_to_lds_b128 %0, %1, off"
               :: "v"(lds_off), "v"(gptr) : "memory");
}
static __device__ __forceinline__ void wait_async0() {
  asm volatile("s_wait_asynccnt 0x0" ::: "memory");
}

// ---------------- K0: f32 -> bf16 bulk convert (x and w1, done ONCE) ----------------
__global__ void cvt_bf16_kernel(const float* __restrict__ src,
                                unsigned short* __restrict__ dst, int n4) {
  int i = blockIdx.x * blockDim.x + threadIdx.x;
  if (i < n4) {
    float4 v = ((const float4*)src)[i];
    uint2 o; o.x = pk2(v.x, v.y); o.y = pk2(v.z, v.w);
    ((uint2*)dst)[i] = o;
  }
}

// ---------------- K1: w2sum[e,h] = sum_d w2[e,h,d] ----------------
__global__ void w2sum_kernel(const float* __restrict__ w2, float* __restrict__ w2s) {
  int lane = threadIdx.x & 31;
  int row  = blockIdx.x * 8 + (threadIdx.x >> 5);      // 0..E*H-1
  const float* src = w2 + (size_t)row * DDIM;
  float acc = 0.f;
  #pragma unroll
  for (int j = 0; j < 16; ++j) acc += src[lane + 32 * j];
  #pragma unroll
  for (int off = 16; off >= 1; off >>= 1) acc += __shfl_xor(acc, off, 32);
  if (lane == 0) w2s[row] = acc;
}

// ---------------- K1b: b2sum[e] = sum_d b2[e,d] ----------------
__global__ void b2sum_kernel(const float* __restrict__ b2, float* __restrict__ b2s) {
  int lane = threadIdx.x & 31;
  int e    = threadIdx.x >> 5;
  float acc = 0.f;
  #pragma unroll
  for (int j = 0; j < 16; ++j) acc += b2[e * DDIM + lane + 32 * j];
  #pragma unroll
  for (int off = 16; off >= 1; off >>= 1) acc += __shfl_xor(acc, off, 32);
  if (lane == 0) b2s[e] = acc;
}

// ---------------- K2: gating (top-2 softmax) + z bias seed ----------------
__global__ void gate_kernel(const float* __restrict__ x,  const float* __restrict__ gw,
                            const float* __restrict__ gb, const float* __restrict__ b2s,
                            float* __restrict__ wgt, float* __restrict__ z) {
  int lane = threadIdx.x & 31;
  int t    = blockIdx.x * 8 + (threadIdx.x >> 5);      // one wave per token
  const float* xr = x + (size_t)t * DDIM;
  float acc[NEXP];
  #pragma unroll
  for (int e = 0; e < NEXP; ++e) acc[e] = 0.f;
  for (int j = 0; j < 16; ++j) {
    int d = lane + 32 * j;
    float xv = xr[d];
    const float4* g4 = (const float4*)(gw + d * NEXP);
    float4 g0 = g4[0], g1 = g4[1];
    acc[0] += xv * g0.x; acc[1] += xv * g0.y; acc[2] += xv * g0.z; acc[3] += xv * g0.w;
    acc[4] += xv * g1.x; acc[5] += xv * g1.y; acc[6] += xv * g1.z; acc[7] += xv * g1.w;
  }
  #pragma unroll
  for (int e = 0; e < NEXP; ++e) {
    #pragma unroll
    for (int off = 16; off >= 1; off >>= 1) acc[e] += __shfl_xor(acc[e], off, 32);
  }
  if (lane == 0) {
    float l[NEXP];
    #pragma unroll
    for (int e = 0; e < NEXP; ++e) l[e] = acc[e] + gb[e];
    int i0 = 0; float v0 = l[0];
    for (int e = 1; e < NEXP; ++e) if (l[e] > v0) { v0 = l[e]; i0 = e; }
    int i1 = (i0 == 0) ? 1 : 0; float v1 = -3.4e38f;
    for (int e = 0; e < NEXP; ++e) if (e != i0 && l[e] > v1) { v1 = l[e]; i1 = e; }
    float p1  = expf(v1 - v0);               // v0 >= v1 so safe
    float inv = 1.f / (1.f + p1);
    float w0 = inv, w1 = p1 * inv;
    #pragma unroll
    for (int e = 0; e < NEXP; ++e) wgt[t * NEXP + e] = 0.f;
    wgt[t * NEXP + i0] = w0;
    wgt[t * NEXP + i1] = w1;
    z[t] = w0 * b2s[i0] + w1 * b2s[i1];      // bias term of collapsed 2nd GEMM
  }
}

// ---------------- K3: main GEMM  z[t] += w[t,e] * gelu(X W1e + b1e) . w2sum[e] ----------------
// LDS: A tile 128KB + B double-buffer 2x16KB = 160KB (2 blocks per 320KB WGP).
__global__ __launch_bounds__(256) void moe_main_kernel(
    const unsigned short* __restrict__ xb,  const unsigned short* __restrict__ w1b,
    const float* __restrict__ b1,  const float* __restrict__ w2s,
    const float* __restrict__ wgt, float* __restrict__ z) {
  extern __shared__ __align__(16) char smem[];
  constexpr int ABYTES = 128 * DDIM * 2;     // 131072
  constexpr int BBYTES = DDIM * 16 * 2;      // 16384
  const uint4* Ash4 = (const uint4*)smem;
  const unsigned ldsBase = (unsigned)(unsigned long long)smem;  // LDS byte offset
  const unsigned aBase = ldsBase;
  const unsigned bBase = ldsBase + ABYTES;

  const int e       = blockIdx.y;
  const int tokBase = blockIdx.x * 128;
  const int tid     = threadIdx.x;
  const int lane    = tid & 31;
  const int wave    = tid >> 5;
  const int halfl   = lane & 15;
  const int g       = lane >> 4;

  // Async-stage A tile once: 128 tokens x 512 D, bf16 (8192 x b128 transfers)
  {
    const char* xg = (const char*)(xb + (size_t)tokBase * DDIM);
    for (int i = tid; i < ABYTES / 16; i += 256)
      async_ld_b128(aBase + (unsigned)i * 16u, xg + (size_t)i * 16);
  }
  // Async-stage B tile hc into buffer buf
  auto stageB = [&](int hc, int buf) {
    unsigned lb = bBase + (unsigned)buf * BBYTES;
    for (int d = tid; d < DDIM; d += 256) {
      const char* gp = (const char*)(w1b + ((size_t)e * DDIM + d) * HDIM + hc * 16);
      unsigned lo = lb + (unsigned)d * 32u;
      async_ld_b128(lo,       gp);
      async_ld_b128(lo + 16u, gp + 16);
    }
  };
  stageB(0, 0);
  wait_async0();
  __syncthreads();

  const int arow = wave * 16 + halfl;     // this wave owns tokens tokBase + wave*16 ..
  float zacc[8];
  #pragma unroll
  for (int r = 0; r < 8; ++r) zacc[r] = 0.f;

  for (int hc = 0; hc < HDIM / 16; ++hc) {
    // prefetch next B slab asynchronously into the other buffer
    if (hc + 1 < HDIM / 16) stageB(hc + 1, (hc + 1) & 1);

    const uint4* Bcur = (const uint4*)(smem + ABYTES + (hc & 1) * BBYTES);
    const int   col = hc * 16 + halfl;      // N column handled by this lane
    const float b1v = b1[e * HDIM + col];
    const float w2v = w2s[e * HDIM + col];

    // Software-pipelined fragment fetch: issue LDS loads for step kc+1 before
    // the WMMA of step kc so the scheduler can use partial dscnt waits.
    v8f acc = {};
    int ai0 = arow * 64 + g;                 // uint4 index (16B units)
    auto bfrag = [&](int kc2) {
      const unsigned short* Bs = (const unsigned short*)Bcur;
      v16bf v;
      #pragma unroll
      for (int s = 0; s < 16; ++s) {
        int k = kc2 * 32 + (s < 8 ? 8 * g + s : 16 + 8 * g + (s - 8));
        v[s] = __builtin_bit_cast(__bf16, Bs[k * 16 + halfl]);
      }
      return v;
    };
    v16bf aCur = frag_from(Ash4, ai0, ai0 + 2);
    v16bf bCur = bfrag(0);
    #pragma unroll
    for (int kc = 0; kc < DDIM / 32; ++kc) {
      v16bf aNxt = aCur, bNxt = bCur;
      if (kc + 1 < DDIM / 32) {
        int ai = arow * 64 + (kc + 1) * 4 + g;
        aNxt = frag_from(Ash4, ai, ai + 2);
        bNxt = bfrag(kc + 1);
      }
      acc = __builtin_amdgcn_wmma_f32_16x16x32_bf16(false, aCur, false, bCur,
                                                    (short)0, acc, false, false);
      aCur = aNxt; bCur = bNxt;
    }

    #pragma unroll
    for (int r = 0; r < 8; ++r) {
      float u = acc[r] + b1v;                         // rows r / r+8, col fixed per lane
      zacc[r] += gelu_exact(u) * w2v;
    }

    wait_async0();        // next buffer landed (this thread's copies)
    __syncthreads();      // ...and everyone else's
  }

  // Reduce across the 16 N-lanes of each half-wave (offsets <=8 stay within halves)
  #pragma unroll
  for (int r = 0; r < 8; ++r) {
    #pragma unroll
    for (int off = 8; off >= 1; off >>= 1) zacc[r] += __shfl_xor(zacc[r], off, 32);
  }

  if (halfl == 0) {                                   // lanes 0 and 16
    int rbase = tokBase + wave * 16 + g * 8;
    #pragma unroll
    for (int r = 0; r < 8; ++r) {
      int t = rbase + r;
      float wv = wgt[t * NEXP + e];
      if (wv != 0.f) atomicAdd(&z[t], wv * zacc[r]);
    }
  }
}

// ---------------- K4: log_softmax over seq dim per batch ----------------
__global__ void lsm_kernel(const float* __restrict__ z, float* __restrict__ out) {
  int b = blockIdx.x, tid = threadIdx.x;
  __shared__ float red[256];
  const float* zb = z + b * 1024;
  float v[4]; float m = -3.4e38f;
  #pragma unroll
  for (int j = 0; j < 4; ++j) { v[j] = zb[tid + 256 * j]; m = fmaxf(m, v[j]); }
  red[tid] = m; __syncthreads();
  for (int s = 128; s >= 1; s >>= 1) {
    if (tid < s) red[tid] = fmaxf(red[tid], red[tid + s]);
    __syncthreads();
  }
  m = red[0]; __syncthreads();
  float se = 0.f;
  #pragma unroll
  for (int j = 0; j < 4; ++j) se += expf(v[j] - m);
  red[tid] = se; __syncthreads();
  for (int s = 128; s >= 1; s >>= 1) {
    if (tid < s) red[tid] += red[tid + s];
    __syncthreads();
  }
  float ls = logf(red[0]);
  #pragma unroll
  for (int j = 0; j < 4; ++j) out[b * 1024 + tid + 256 * j] = v[j] - m - ls;
}

// ---------------- launch ----------------
extern "C" void kernel_launch(void* const* d_in, const int* in_sizes, int n_in,
                              void* d_out, int out_size, void* d_ws, size_t ws_size,
                              hipStream_t stream) {
  const float* x  = (const float*)d_in[0];
  const float* gw = (const float*)d_in[1];
  const float* gb = (const float*)d_in[2];
  const float* w1 = (const float*)d_in[3];
  const float* b1 = (const float*)d_in[4];
  const float* w2 = (const float*)d_in[5];
  const float* b2 = (const float*)d_in[6];
  float* out = (float*)d_out;

  char*  ws  = (char*)d_ws;
  float* wgt = (float*)(ws);                          // [T, E]  = 262144 B
  float* z   = (float*)(ws + 262144);                 // [T]     =  32768 B
  float* w2s = (float*)(ws + 294912);                 // [E, H]  =  65536 B
  float* b2s = (float*)(ws + 360448);                 // [E]     =     32 B
  unsigned short* xb  = (unsigned short*)(ws + 393216);             // 8.4 MB
  unsigned short* w1b = (unsigned short*)(ws + 393216 + 8388608);   // 16.8 MB

  // one-time bf16 conversions (removes per-block redundant cvt VALU + halves L2 traffic)
  cvt_bf16_kernel<<<(T_TOK * DDIM / 4) / 256, 256, 0, stream>>>(x, xb, T_TOK * DDIM / 4);
  cvt_bf16_kernel<<<(NEXP * DDIM * HDIM / 4) / 256, 256, 0, stream>>>(w1, w1b,
                                                                      NEXP * DDIM * HDIM / 4);

  w2sum_kernel<<<(NEXP * HDIM) / 8, 256, 0, stream>>>(w2, w2s);
  b2sum_kernel<<<1, 256, 0, stream>>>(b2, b2s);
  gate_kernel<<<T_TOK / 8, 256, 0, stream>>>(x, gw, gb, b2s, wgt, z);

  dim3 grid(T_TOK / 128, NEXP);
  size_t lds = (size_t)(128 * DDIM * 2 + 2 * DDIM * 16 * 2);   // 160 KB
  moe_main_kernel<<<grid, 256, lds, stream>>>(xb, w1b, b1, w2s, wgt, z);

  lsm_kernel<<<8, 256, 0, stream>>>(z, out);
}
